// SA_DGNet_5274219839865
// MI455X (gfx1250) — compile-verified, matrix operand order checked by reading, not run on hardware
//
#include <hip/hip_runtime.h>
#include <cstdint>
#include <cstddef>

#define N_ROWS 8192
#define D_IN   1024
#define D_H    128
#define KSLAB  32
#define NSLAB  (D_IN / KSLAB)   // 32
#define GROWP  40                // GLU LDS row stride in halves (80 B, breaks bank conflicts)
#define AROWP  136               // attn LDS row stride in halves (272 B)

typedef __attribute__((ext_vector_type(16))) __bf16        v16bf;
typedef __attribute__((ext_vector_type(8)))  float         v8f;
typedef __attribute__((ext_vector_type(4)))  unsigned int  uint4v;

union Frag {
  v16bf v;
  uint4v q[2];
  unsigned short us[16];
};

__device__ __forceinline__ unsigned short f32_bf16(float f) {
  unsigned int u = __float_as_uint(f);
  u += 0x7FFFu + ((u >> 16) & 1u);   // round-to-nearest-even
  return (unsigned short)(u >> 16);
}

// CDNA5 async memory->LDS copy: 16 bytes per lane, tracked by ASYNCcnt.
__device__ __forceinline__ void async_cp16(unsigned lds_off,
                                           const unsigned short* gaddr) {
  asm volatile("global_load_async_to_lds_b128 %0, %1, off"
               :
               : "v"(lds_off), "v"((unsigned long long)(uintptr_t)gaddr)
               : "memory");
}
__device__ __forceinline__ void wait_async0() {
  asm volatile("s_wait_asynccnt 0x0" ::: "memory");
}

// ---------------------------------------------------------------- converts
__global__ void cvt_bf16_kernel(const float* __restrict__ src,
                                unsigned short* __restrict__ dst, int n) {
  for (int i = blockIdx.x * blockDim.x + threadIdx.x; i < n;
       i += gridDim.x * blockDim.x)
    dst[i] = f32_bf16(src[i]);
}

// W [K][Nc] f32  ->  Wt [Nc][K] bf16   (so B-fragments read contiguously along K)
__global__ void cvt_bf16_T_kernel(const float* __restrict__ src,
                                  unsigned short* __restrict__ dst,
                                  int K, int Nc) {
  int n = K * Nc;
  for (int i = blockIdx.x * blockDim.x + threadIdx.x; i < n;
       i += gridDim.x * blockDim.x) {
    int r = i / Nc, c = i - r * Nc;
    dst[(size_t)c * K + r] = f32_bf16(src[i]);
  }
}

// ---------------------------------------------------------------- GLU
// h = (x@Wfc + bfc) * sigmoid(x@Wg + bg); async double-buffered LDS pipeline.
__global__ void __launch_bounds__(256)
__attribute__((amdgpu_waves_per_eu(1))) glu_kernel(
    const unsigned short* __restrict__ xb,
    const unsigned short* __restrict__ WfcT,   // [128][1024]
    const float* __restrict__ bfc,
    const unsigned short* __restrict__ WgT,    // [128][1024]
    const float* __restrict__ bg,
    unsigned short* __restrict__ hb) {
  __shared__ unsigned short Xs[2][128 * GROWP];  // 2 x 10 KB
  __shared__ unsigned short Wf[2][128 * GROWP];  // 2 x 10 KB
  __shared__ unsigned short Wg[2][128 * GROWP];  // 2 x 10 KB

  const int lane = threadIdx.x & 31;
  const int wid  = threadIdx.x >> 5;
  const int lo   = lane & 15;
  const int hi   = lane >> 4;
  const int row0 = blockIdx.x * 128 + wid * 16;
  const unsigned short* xrow0 = xb + (size_t)blockIdx.x * 128 * D_IN;

  v8f accF[8], accG[8];
#pragma unroll
  for (int t = 0; t < 8; ++t) { accF[t] = {}; accG[t] = {}; }

  // Issue one 32-wide K-slab (x tile + both weight slabs) as async LDS copies:
  // 512 16B-chunks per array, 2 per thread per array -> 6 async ops/thread.
  auto issue_slab = [&](int s, int buf) {
    const int k0 = s * KSLAB;
    const unsigned xo = (unsigned)(uintptr_t)&Xs[buf][0];
    const unsigned fo = (unsigned)(uintptr_t)&Wf[buf][0];
    const unsigned go = (unsigned)(uintptr_t)&Wg[buf][0];
    for (int c = threadIdx.x; c < 512; c += 256) {
      const int row = c >> 2, part = c & 3;
      const unsigned loff = (unsigned)(row * GROWP + part * 8) * 2u;
      const size_t gidx = (size_t)row * D_IN + k0 + part * 8;
      async_cp16(xo + loff, xrow0 + gidx);
      async_cp16(fo + loff, WfcT + gidx);
      async_cp16(go + loff, WgT + gidx);
    }
  };

  issue_slab(0, 0);
  for (int s = 0; s < NSLAB; ++s) {
    const int cur = s & 1;
    wait_async0();       // slab s landed in LDS
    __syncthreads();     // visible to all waves; prior slab fully consumed
    if (s + 1 < NSLAB) issue_slab(s + 1, cur ^ 1);

    const unsigned short* xs  = &Xs[cur][0];
    const unsigned short* wfs = &Wf[cur][0];
    const unsigned short* wgs = &Wg[cur][0];

    Frag a;
    const int ar = wid * 16 + lo;
    a.q[0] = *reinterpret_cast<const uint4v*>(xs + ar * GROWP + hi * 8);
    a.q[1] = *reinterpret_cast<const uint4v*>(xs + ar * GROWP + 16 + hi * 8);
#pragma unroll
    for (int g = 0; g < 2; ++g) {
      Frag bF[4], bG[4];
#pragma unroll
      for (int j = 0; j < 4; ++j) {
        const int n = (g * 4 + j) * 16 + lo;
        bF[j].q[0] = *reinterpret_cast<const uint4v*>(wfs + n * GROWP + hi * 8);
        bF[j].q[1] = *reinterpret_cast<const uint4v*>(wfs + n * GROWP + 16 + hi * 8);
        bG[j].q[0] = *reinterpret_cast<const uint4v*>(wgs + n * GROWP + hi * 8);
        bG[j].q[1] = *reinterpret_cast<const uint4v*>(wgs + n * GROWP + 16 + hi * 8);
      }
#pragma unroll
      for (int j = 0; j < 4; ++j) {
        accF[g * 4 + j] = __builtin_amdgcn_wmma_f32_16x16x32_bf16(
            false, a.v, false, bF[j].v, (short)0, accF[g * 4 + j], false, false);
        accG[g * 4 + j] = __builtin_amdgcn_wmma_f32_16x16x32_bf16(
            false, a.v, false, bG[j].v, (short)0, accG[g * 4 + j], false, false);
      }
    }
    __syncthreads();     // done reading buf[cur] before slab s+2 overwrites it
  }

#pragma unroll
  for (int t = 0; t < 8; ++t) {
    const int n = t * 16 + lo;
    const float bfv = bfc[n], bgv = bg[n];
#pragma unroll
    for (int r = 0; r < 8; ++r) {
      float f = accF[t][r] + bfv;
      float g = accG[t][r] + bgv;
      float h = f * (1.0f / (1.0f + __expf(-g)));
      int m = row0 + r + 8 * hi;
      hb[(size_t)m * D_H + n] = f32_bf16(h);
    }
  }
}

// ---------------------------------------------------------------- QKV
__global__ void __launch_bounds__(256)
__attribute__((amdgpu_waves_per_eu(1))) qkv_kernel(
    const unsigned short* __restrict__ hb,
    const unsigned short* __restrict__ WqT, const float* __restrict__ bq,
    const unsigned short* __restrict__ WkT, const float* __restrict__ bk,
    const unsigned short* __restrict__ WvT, const float* __restrict__ bv,
    unsigned short* __restrict__ qb,
    unsigned short* __restrict__ kb,
    unsigned short* __restrict__ vb) {
  const int lane = threadIdx.x & 31;
  const int wid  = threadIdx.x >> 5;
  const int lo   = lane & 15;
  const int hi   = lane >> 4;
  const int row0 = blockIdx.x * 128 + wid * 16;

  const unsigned short* arow = hb + (size_t)(row0 + lo) * D_H;
  Frag a[4];
#pragma unroll
  for (int kk = 0; kk < 4; ++kk) {
    a[kk].q[0] = *reinterpret_cast<const uint4v*>(arow + kk * 32 + hi * 8);
    a[kk].q[1] = *reinterpret_cast<const uint4v*>(arow + kk * 32 + 16 + hi * 8);
  }

#pragma unroll
  for (int t = 0; t < 8; ++t) {
    const int n = t * 16 + lo;
    Frag bQ[4], bK[4], bV[4];
#pragma unroll
    for (int kk = 0; kk < 4; ++kk) {
      const unsigned short* wq = WqT + (size_t)n * D_H + kk * 32;
      const unsigned short* wk = WkT + (size_t)n * D_H + kk * 32;
      const unsigned short* wv = WvT + (size_t)n * D_H + kk * 32;
      bQ[kk].q[0] = *reinterpret_cast<const uint4v*>(wq + hi * 8);
      bQ[kk].q[1] = *reinterpret_cast<const uint4v*>(wq + 16 + hi * 8);
      bK[kk].q[0] = *reinterpret_cast<const uint4v*>(wk + hi * 8);
      bK[kk].q[1] = *reinterpret_cast<const uint4v*>(wk + 16 + hi * 8);
      bV[kk].q[0] = *reinterpret_cast<const uint4v*>(wv + hi * 8);
      bV[kk].q[1] = *reinterpret_cast<const uint4v*>(wv + 16 + hi * 8);
    }
    v8f aq = {}, ak = {}, av = {};
#pragma unroll
    for (int kk = 0; kk < 4; ++kk) {
      aq = __builtin_amdgcn_wmma_f32_16x16x32_bf16(false, a[kk].v, false, bQ[kk].v,
                                                   (short)0, aq, false, false);
      ak = __builtin_amdgcn_wmma_f32_16x16x32_bf16(false, a[kk].v, false, bK[kk].v,
                                                   (short)0, ak, false, false);
      av = __builtin_amdgcn_wmma_f32_16x16x32_bf16(false, a[kk].v, false, bV[kk].v,
                                                   (short)0, av, false, false);
    }
    const float bqv = bq[n], bkv = bk[n], bvv = bv[n];
#pragma unroll
    for (int r = 0; r < 8; ++r) {
      int m = row0 + r + 8 * hi;
      qb[(size_t)m * D_H + n] = f32_bf16(aq[r] + bqv);
      kb[(size_t)m * D_H + n] = f32_bf16(ak[r] + bkv);
      vb[(size_t)m * D_H + n] = f32_bf16(av[r] + bvv);
    }
  }
}

// ---------------------------------------------------------------- flash attention + out proj
__global__ void __launch_bounds__(256)
__attribute__((amdgpu_waves_per_eu(1))) attn_kernel(
    const unsigned short* __restrict__ qb,
    const unsigned short* __restrict__ kb,
    const unsigned short* __restrict__ vb,
    const float* __restrict__ Wout,
    const float* __restrict__ bout,
    float* __restrict__ out) {
  __shared__ unsigned short Kls[2][32 * AROWP];  // 2 x 8.5 KB (async-filled)
  __shared__ unsigned short Vls[2][32 * AROWP];  // 2 x 8.5 KB (async-filled)
  __shared__ unsigned short Plds[8][16 * 32];    // 8 KB: per-wave P relayout
  const int lane = threadIdx.x & 31;
  const int wid  = threadIdx.x >> 5;
  const int lo   = lane & 15;
  const int hi   = lane >> 4;
  const int row0 = blockIdx.x * 128 + wid * 16;

  // Q A-fragments, resident for the whole kernel
  Frag qf[4];
  const unsigned short* qrow = qb + (size_t)(row0 + lo) * D_H;
#pragma unroll
  for (int kk = 0; kk < 4; ++kk) {
    qf[kk].q[0] = *reinterpret_cast<const uint4v*>(qrow + kk * 32 + hi * 8);
    qf[kk].q[1] = *reinterpret_cast<const uint4v*>(qrow + kk * 32 + 16 + hi * 8);
  }

  v8f o[8];
#pragma unroll
  for (int t = 0; t < 8; ++t) o[t] = {};
  float mrow[8], lrow[8];
#pragma unroll
  for (int r = 0; r < 8; ++r) { mrow[r] = -3.0e38f; lrow[r] = 0.0f; }

  const float scale = 0.0883883476483184f;  // 1/sqrt(128)

  // async stage of one 32-row K tile + V tile: 512 chunks/array, 2/thread/array
  auto issue_tile = [&](int kv, int buf) {
    const unsigned ko = (unsigned)(uintptr_t)&Kls[buf][0];
    const unsigned vo = (unsigned)(uintptr_t)&Vls[buf][0];
    for (int c = threadIdx.x; c < 512; c += 256) {
      const int row = c >> 4, part = c & 15;
      const unsigned loff = (unsigned)(row * AROWP + part * 8) * 2u;
      const size_t gidx = (size_t)(kv + row) * D_H + part * 8;
      async_cp16(ko + loff, kb + gidx);
      async_cp16(vo + loff, vb + gidx);
    }
  };

  issue_tile(0, 0);
  for (int kv0 = 0; kv0 < N_ROWS; kv0 += 32) {
    const int cur = (kv0 >> 5) & 1;
    wait_async0();       // tile kv0 landed
    __syncthreads();     // visible block-wide; previous tile fully consumed
    if (kv0 + 32 < N_ROWS) issue_tile(kv0 + 32, cur ^ 1);  // overlap next copy

    const unsigned short* kt = &Kls[cur][0];
    const unsigned short* vt = &Vls[cur][0];

    // scores S = q @ k^T for 16 query rows x 32 keys (two 16x16 tiles)
    Frag b0[4], b1[4];
#pragma unroll
    for (int kk = 0; kk < 4; ++kk) {
      b0[kk].q[0] = *reinterpret_cast<const uint4v*>(kt + lo * AROWP + kk * 32 + hi * 8);
      b0[kk].q[1] = *reinterpret_cast<const uint4v*>(kt + lo * AROWP + kk * 32 + 16 + hi * 8);
      b1[kk].q[0] = *reinterpret_cast<const uint4v*>(kt + (16 + lo) * AROWP + kk * 32 + hi * 8);
      b1[kk].q[1] = *reinterpret_cast<const uint4v*>(kt + (16 + lo) * AROWP + kk * 32 + 16 + hi * 8);
    }
    v8f s0 = {}, s1 = {};
#pragma unroll
    for (int kk = 0; kk < 4; ++kk) {
      s0 = __builtin_amdgcn_wmma_f32_16x16x32_bf16(false, qf[kk].v, false, b0[kk].v,
                                                   (short)0, s0, false, false);
      s1 = __builtin_amdgcn_wmma_f32_16x16x32_bf16(false, qf[kk].v, false, b1[kk].v,
                                                   (short)0, s1, false, false);
    }

    // online softmax; rows of a 16x16 D-tile live across lanes 0..15 / 16..31
    float alpha[8];
#pragma unroll
    for (int r = 0; r < 8; ++r) {
      float a = s0[r] * scale, b = s1[r] * scale;
      float t = fmaxf(a, b);
      t = fmaxf(t, __shfl_xor(t, 1, 32));
      t = fmaxf(t, __shfl_xor(t, 2, 32));
      t = fmaxf(t, __shfl_xor(t, 4, 32));
      t = fmaxf(t, __shfl_xor(t, 8, 32));
      float mn = fmaxf(mrow[r], t);
      alpha[r] = __expf(mrow[r] - mn);
      mrow[r] = mn;
      float p0 = __expf(a - mn);
      float p1 = __expf(b - mn);
      s0[r] = p0; s1[r] = p1;
      float rs = p0 + p1;
      rs += __shfl_xor(rs, 1, 32);
      rs += __shfl_xor(rs, 2, 32);
      rs += __shfl_xor(rs, 4, 32);
      rs += __shfl_xor(rs, 8, 32);
      lrow[r] = lrow[r] * alpha[r] + rs;
    }

    // P: C/D layout -> A layout through per-wave LDS region (bf16)
    unsigned short* pw = Plds[wid];
#pragma unroll
    for (int r = 0; r < 8; ++r) {
      int m = r + 8 * hi;
      pw[m * 32 + lo]      = f32_bf16(s0[r]);
      pw[m * 32 + 16 + lo] = f32_bf16(s1[r]);
    }
    __builtin_amdgcn_wave_barrier();
    Frag pf;
    pf.q[0] = *reinterpret_cast<const uint4v*>(pw + lo * 32 + hi * 8);
    pf.q[1] = *reinterpret_cast<const uint4v*>(pw + lo * 32 + 16 + hi * 8);

    // rescale O, then O += P @ V
#pragma unroll
    for (int t = 0; t < 8; ++t)
#pragma unroll
      for (int r = 0; r < 8; ++r) o[t][r] *= alpha[r];

#pragma unroll
    for (int t = 0; t < 8; ++t) {
      Frag bvf;
#pragma unroll
      for (int e = 0; e < 16; ++e) {
        int krow = ((e >> 3) * 16) + hi * 8 + (e & 7);
        bvf.us[e] = vt[krow * AROWP + t * 16 + lo];
      }
      o[t] = __builtin_amdgcn_wmma_f32_16x16x32_bf16(false, pf.v, false, bvf.v,
                                                     (short)0, o[t], false, false);
    }
  }

  // epilogue: out = (O / l) @ W_out + b_out, fused lane reduction
  float acc[8];
#pragma unroll
  for (int r = 0; r < 8; ++r) acc[r] = 0.0f;
#pragma unroll
  for (int t = 0; t < 8; ++t) {
    float w = Wout[t * 16 + lo];
#pragma unroll
    for (int r = 0; r < 8; ++r) acc[r] += o[t][r] * w;
  }
#pragma unroll
  for (int r = 0; r < 8; ++r) {
    float v = acc[r] / lrow[r];
    v += __shfl_xor(v, 1, 32);
    v += __shfl_xor(v, 2, 32);
    v += __shfl_xor(v, 4, 32);
    v += __shfl_xor(v, 8, 32);
    if (lo == 0) out[row0 + r + 8 * hi] = v + bout[0];
  }
}

// ---------------------------------------------------------------- launch
extern "C" void kernel_launch(void* const* d_in, const int* in_sizes, int n_in,
                              void* d_out, int out_size, void* d_ws, size_t ws_size,
                              hipStream_t stream) {
  (void)in_sizes; (void)n_in; (void)out_size; (void)ws_size;
  const float* x     = (const float*)d_in[0];
  const float* W_fc  = (const float*)d_in[1];
  const float* b_fc  = (const float*)d_in[2];
  const float* W_g   = (const float*)d_in[3];
  const float* b_g   = (const float*)d_in[4];
  const float* W_q   = (const float*)d_in[5];
  const float* b_q   = (const float*)d_in[6];
  const float* W_k   = (const float*)d_in[7];
  const float* b_k   = (const float*)d_in[8];
  const float* W_v   = (const float*)d_in[9];
  const float* b_v   = (const float*)d_in[10];
  const float* W_out = (const float*)d_in[11];
  const float* b_out = (const float*)d_in[12];
  float* out = (float*)d_out;

  char* ws = (char*)d_ws;
  size_t off = 0;
  auto alloc = [&](size_t bytes) {
    void* p = ws + off;
    off = (off + bytes + 255) & ~(size_t)255;
    return p;
  };
  unsigned short* xb   = (unsigned short*)alloc((size_t)N_ROWS * D_IN * 2);
  unsigned short* WfcT = (unsigned short*)alloc((size_t)D_IN * D_H * 2);
  unsigned short* WgT  = (unsigned short*)alloc((size_t)D_IN * D_H * 2);
  unsigned short* WqT  = (unsigned short*)alloc((size_t)D_H * D_H * 2);
  unsigned short* WkT  = (unsigned short*)alloc((size_t)D_H * D_H * 2);
  unsigned short* WvT  = (unsigned short*)alloc((size_t)D_H * D_H * 2);
  unsigned short* hb   = (unsigned short*)alloc((size_t)N_ROWS * D_H * 2);
  unsigned short* qb   = (unsigned short*)alloc((size_t)N_ROWS * D_H * 2);
  unsigned short* kb   = (unsigned short*)alloc((size_t)N_ROWS * D_H * 2);
  unsigned short* vb   = (unsigned short*)alloc((size_t)N_ROWS * D_H * 2);

  cvt_bf16_kernel<<<1024, 256, 0, stream>>>(x, xb, N_ROWS * D_IN);
  cvt_bf16_T_kernel<<<256, 256, 0, stream>>>(W_fc, WfcT, D_IN, D_H);
  cvt_bf16_T_kernel<<<256, 256, 0, stream>>>(W_g,  WgT,  D_IN, D_H);
  cvt_bf16_T_kernel<<<32, 256, 0, stream>>>(W_q, WqT, D_H, D_H);
  cvt_bf16_T_kernel<<<32, 256, 0, stream>>>(W_k, WkT, D_H, D_H);
  cvt_bf16_T_kernel<<<32, 256, 0, stream>>>(W_v, WvT, D_H, D_H);

  glu_kernel<<<64, 256, 0, stream>>>(xb, WfcT, b_fc, WgT, b_g, hb);
  qkv_kernel<<<64, 256, 0, stream>>>(hb, WqT, b_q, WkT, b_k, WvT, b_v, qb, kb, vb);
  attn_kernel<<<64, 256, 0, stream>>>(qb, kb, vb, W_out, b_out, out);
}
